// SelfAttention_30425548325218
// MI455X (gfx1250) — matrix operand order repaired and checked by reading, not compiled, not run on hardware
//
#include <hip/hip_runtime.h>
#include <hip/hip_bf16.h>
#include <math.h>

#define Bsz 4
#define Lsz 2048
#define Esz 1024
#define Hsz 16
#define Dsz 64

typedef __attribute__((ext_vector_type(16))) _Float16 v16h;
typedef __attribute__((ext_vector_type(8)))  _Float16 v8h;
typedef __attribute__((ext_vector_type(8)))  float    v8f;

union AF { v16h v; v8h p[2]; };

// WMMA f32 = f16 x f16 + f32 (16x16x32)
__device__ inline v8f wmma32(v16h a, v16h b, v8f c) {
    return __builtin_amdgcn_wmma_f32_16x16x32_f16(
        false, a, false, b, (short)0, c, false, false);
}

// A-fragment (16x32 f16, row-major, row stride lda halves).
// lane 0-15: M=lane, K 0..15 ; lane 16-31: M=lane-16, K 16..31 (contiguous halves)
__device__ inline v16h load_af(const _Float16* base, int lda) {
    int lane = threadIdx.x & 31;
    const _Float16* p = base + (lane & 15) * lda + ((lane >> 4) << 4);
    AF u;
    u.p[0] = *(const v8h*)(p);
    u.p[1] = *(const v8h*)(p + 8);
    return u.v;
}

// B-fragment (32x16 f16) loaded from transposed storage Bt[N][K] (row stride ldb).
// lane 0-15: N=lane, K 0..15 ; lane 16-31: N=lane-16, K 16..31 (contiguous halves)
__device__ inline v16h load_bf(const _Float16* bt, int ldb) {
    int lane = threadIdx.x & 31;
    const _Float16* p = bt + (lane & 15) * ldb + ((lane >> 4) << 4);
    AF u;
    u.p[0] = *(const v8h*)(p);
    u.p[1] = *(const v8h*)(p + 8);
    return u.v;
}

// CDNA5 async global->LDS DMA (ASYNCcnt path, ISA 15.18 op 98). ldsoff is the
// workgroup-relative LDS byte offset (low 32 bits of a generic shared pointer).
__device__ inline void async_g2l_b128(unsigned ldsoff, const void* gptr) {
    asm volatile("global_load_async_to_lds_b128 %0, %1, off"
                 :: "v"(ldsoff), "v"(gptr) : "memory");
}
__device__ inline void wait_async3() { asm volatile("s_wait_asynccnt 0x3" ::: "memory"); }
__device__ inline void wait_async0() { asm volatile("s_wait_asynccnt 0x0" ::: "memory"); }

__device__ inline float hmax16(float x) {
    #pragma unroll
    for (int m = 1; m < 16; m <<= 1) x = fmaxf(x, __shfl_xor(x, m, 32));
    return x;
}
__device__ inline float hsum16(float x) {
    #pragma unroll
    for (int m = 1; m < 16; m <<= 1) x += __shfl_xor(x, m, 32);
    return x;
}

// ---------------- prep kernels ----------------
__global__ __launch_bounds__(256) void cvt_f32_f16(const float* __restrict__ s,
                                                   _Float16* __restrict__ d, int n) {
    int i = blockIdx.x * 256 + threadIdx.x;
    if (i < n) d[i] = (_Float16)s[i];
}

// src[rows][cols] f32 -> dst[cols][rows] f16
__global__ __launch_bounds__(256) void transpose_to_f16(const float* __restrict__ s,
                                                        _Float16* __restrict__ d,
                                                        int rows, int cols) {
    int i = blockIdx.x * 256 + threadIdx.x;
    if (i < rows * cols) {
        int r = i / cols, c = i % cols;
        d[(size_t)c * rows + r] = (_Float16)s[i];
    }
}

// ---------------- QKV projection ----------------
// C[8192 x 3072] = Xh[8192 x 1024] @ W ; Wt is [3072][1024] f16 (pre-transposed)
// Block: 128 rows x 64 cols, 8 waves. Double-buffered async LDS staging.
// Scatter into Qh,Kh [b*16+h][L][D] and Vt [b*16+h][D][L], all f16.
__global__ __launch_bounds__(256) void qkv_gemm(const _Float16* __restrict__ Xh,
                                                const _Float16* __restrict__ Wt,
                                                const float* __restrict__ bias,
                                                _Float16* __restrict__ Qh,
                                                _Float16* __restrict__ Kh,
                                                _Float16* __restrict__ Vt) {
    // LDS: A buffers 2x(128x32 f16)=2x8KB @0, B buffers 2x(64x32 f16)=2x4KB @16KB
    __shared__ _Float16 smem[12288];
    unsigned sbase = (unsigned)(size_t)(void*)smem;
    int tid = threadIdx.x;
    int wave = tid >> 5, lane = tid & 31;
    int nn = lane & 15, half = lane >> 4;
    int rt = blockIdx.x / 48, ct = blockIdx.x % 48;
    int row0 = rt * 128, col0 = ct * 64;
    const _Float16* Ab = Xh + (size_t)row0 * Esz;
    const _Float16* Bb = Wt + (size_t)col0 * Esz;

    v8f acc[4];
    #pragma unroll
    for (int nt = 0; nt < 4; nt++) {
        float bv = bias[col0 + nt * 16 + nn];
        #pragma unroll
        for (int r = 0; r < 8; r++) acc[nt][r] = bv;
    }

    auto stage = [&](int kc, int buf) {
        const _Float16* Ag = Ab + kc * 32;
        const _Float16* Bg = Bb + kc * 32;
        unsigned abase = sbase + (unsigned)buf * 8192u;
        unsigned bbase = sbase + 16384u + (unsigned)buf * 4096u;
        int u0 = tid * 2, u1 = u0 + 1;
        async_g2l_b128(abase + u0 * 16, Ag + (size_t)(u0 >> 2) * Esz + (u0 & 3) * 8);
        async_g2l_b128(abase + u1 * 16, Ag + (size_t)(u1 >> 2) * Esz + (u1 & 3) * 8);
        async_g2l_b128(bbase + tid * 16, Bg + (size_t)(tid >> 2) * Esz + (tid & 3) * 8);
    };

    stage(0, 0);
    for (int kc = 0; kc < Esz / 32; kc++) {
        int buf = kc & 1;
        if (kc + 1 < Esz / 32) { stage(kc + 1, buf ^ 1); wait_async3(); }
        else                   { wait_async0(); }
        __syncthreads();
        const _Float16* At = smem + buf * 4096;          // halves
        const _Float16* Bt = smem + 8192 + buf * 2048;   // halves
        v16h af = load_af(At + wave * 16 * 32, 32);
        #pragma unroll
        for (int nt = 0; nt < 4; nt++) {
            v16h bf = load_bf(Bt + nt * 16 * 32, 32);
            acc[nt] = wmma32(af, bf, acc[nt]);
        }
        __syncthreads();
    }

    int hh = col0 / 192;
    int which = (col0 % 192) / 64;          // 0=Q 1=K 2=V (64-col tile never straddles)
    #pragma unroll
    for (int nt = 0; nt < 4; nt++) {
        int d = nt * 16 + nn;
        #pragma unroll
        for (int r = 0; r < 8; r++) {
            int g = row0 + wave * 16 + r + 8 * half;
            int bb2 = g >> 11, lq = g & (Lsz - 1);
            _Float16 val = (_Float16)acc[nt][r];
            size_t bh = (size_t)(bb2 * Hsz + hh);
            if (which == 0)      Qh[(bh * Lsz + lq) * Dsz + d] = val;
            else if (which == 1) Kh[(bh * Lsz + lq) * Dsz + d] = val;
            else                 Vt[(bh * Dsz + d) * Lsz + lq] = val;
        }
    }
}

// ---------------- flash attention with ALiBi ----------------
__global__ __launch_bounds__(256) void attn_kernel(const _Float16* __restrict__ Qh,
                                                   const _Float16* __restrict__ Kh,
                                                   const _Float16* __restrict__ Vt,
                                                   _Float16* __restrict__ attnH) {
    __shared__ _Float16 lds[8 * 16 * 32];   // per-wave 16x32 P tile
    int wave = threadIdx.x >> 5;
    int lane = threadIdx.x & 31;
    int t = blockIdx.x * 8 + wave;          // 8192 tiles
    int qt = t & 127, h = (t >> 7) & 15, b = t >> 11;
    int q0 = qt * 16;
    int nn = lane & 15, half = lane >> 4;
    const float NEG_INF = -__builtin_inff();

    size_t bh = (size_t)(b * Hsz + h);
    const _Float16* Qb = Qh + (bh * Lsz + q0) * Dsz;
    const _Float16* Kb = Kh + bh * Lsz * Dsz;
    const _Float16* Vb = Vt + bh * Dsz * Lsz;

    v16h qf0 = load_af(Qb, Dsz);            // d 0..31
    v16h qf1 = load_af(Qb + 32, Dsz);       // d 32..63

    float slope = exp2f(-(1.0f + 7.0f * (float)h / 15.0f));
    float m[8], l[8];
    v8f o[4];
    #pragma unroll
    for (int r = 0; r < 8; r++) { m[r] = NEG_INF; l[r] = 0.0f; }
    #pragma unroll
    for (int nt = 0; nt < 4; nt++)
        #pragma unroll
        for (int r = 0; r < 8; r++) o[nt][r] = 0.0f;

    _Float16* pl = lds + wave * 512;
    int jend = (q0 + 15) >> 5;
    for (int j = 0; j <= jend; j++) {
        int k0 = j * 32;
        v8f s0, s1;
        #pragma unroll
        for (int r = 0; r < 8; r++) { s0[r] = 0.0f; s1[r] = 0.0f; }
        v16h kf;
        kf = load_bf(Kb + (size_t)k0 * Dsz, Dsz);             s0 = wmma32(qf0, kf, s0);
        kf = load_bf(Kb + (size_t)k0 * Dsz + 32, Dsz);        s0 = wmma32(qf1, kf, s0);
        kf = load_bf(Kb + (size_t)(k0 + 16) * Dsz, Dsz);      s1 = wmma32(qf0, kf, s1);
        kf = load_bf(Kb + (size_t)(k0 + 16) * Dsz + 32, Dsz); s1 = wmma32(qf1, kf, s1);

        #pragma unroll
        for (int r = 0; r < 8; r++) {
            int qrow = q0 + r + 8 * half;
            int key0 = k0 + nn;
            int key1 = k0 + 16 + nn;
            float v0 = (key0 <= qrow) ? s0[r] * 0.125f + slope * (float)(key0 - qrow)
                                      : NEG_INF;
            float v1 = (key1 <= qrow) ? s1[r] * 0.125f + slope * (float)(key1 - qrow)
                                      : NEG_INF;
            float rmax = hmax16(fmaxf(v0, v1));
            float mnew = fmaxf(m[r], rmax);
            float corr = __expf(m[r] - mnew);
            float p0 = __expf(v0 - mnew);
            float p1 = __expf(v1 - mnew);
            l[r] = l[r] * corr + hsum16(p0 + p1);
            m[r] = mnew;
            o[0][r] *= corr; o[1][r] *= corr; o[2][r] *= corr; o[3][r] *= corr;
            int M = r + 8 * half;
            pl[M * 32 + nn]      = (_Float16)p0;
            pl[M * 32 + 16 + nn] = (_Float16)p1;
        }
        v16h pf = load_af(pl, 32);          // same-wave DS RAW: DScnt wait by compiler
        #pragma unroll
        for (int nt = 0; nt < 4; nt++) {
            v16h vf = load_bf(Vb + (size_t)(nt * 16) * Lsz + k0, Lsz);
            o[nt] = wmma32(pf, vf, o[nt]);
        }
    }

    #pragma unroll
    for (int r = 0; r < 8; r++) {
        float inv = 1.0f / l[r];
        int qrow = q0 + r + 8 * half;
        size_t base = ((size_t)(b * Lsz + qrow)) * (Hsz * Dsz) + h * Dsz;
        #pragma unroll
        for (int nt = 0; nt < 4; nt++)
            attnH[base + nt * 16 + nn] = (_Float16)(o[nt][r] * inv);
    }
}

// ---------------- output projection ----------------
// out[8192 x 1024] f32 = attnH[8192 x 1024] f16 @ W_out ; Wt [1024][1024] f16
__global__ __launch_bounds__(256) void out_gemm(const _Float16* __restrict__ Ah,
                                                const _Float16* __restrict__ Wt,
                                                const float* __restrict__ bias,
                                                float* __restrict__ out) {
    __shared__ _Float16 smem[12288];
    unsigned sbase = (unsigned)(size_t)(void*)smem;
    int tid = threadIdx.x;
    int wave = tid >> 5, lane = tid & 31;
    int nn = lane & 15, half = lane >> 4;
    int rt = blockIdx.x / 16, ct = blockIdx.x % 16;
    int row0 = rt * 128, col0 = ct * 64;
    const int K = Hsz * Dsz;
    const _Float16* Ab = Ah + (size_t)row0 * K;
    const _Float16* Bb = Wt + (size_t)col0 * K;

    v8f acc[4];
    #pragma unroll
    for (int nt = 0; nt < 4; nt++) {
        float bv = bias[col0 + nt * 16 + nn];
        #pragma unroll
        for (int r = 0; r < 8; r++) acc[nt][r] = bv;
    }

    auto stage = [&](int kc, int buf) {
        const _Float16* Ag = Ab + kc * 32;
        const _Float16* Bg = Bb + kc * 32;
        unsigned abase = sbase + (unsigned)buf * 8192u;
        unsigned bbase = sbase + 16384u + (unsigned)buf * 4096u;
        int u0 = tid * 2, u1 = u0 + 1;
        async_g2l_b128(abase + u0 * 16, Ag + (size_t)(u0 >> 2) * K + (u0 & 3) * 8);
        async_g2l_b128(abase + u1 * 16, Ag + (size_t)(u1 >> 2) * K + (u1 & 3) * 8);
        async_g2l_b128(bbase + tid * 16, Bg + (size_t)(tid >> 2) * K + (tid & 3) * 8);
    };

    stage(0, 0);
    for (int kc = 0; kc < K / 32; kc++) {
        int buf = kc & 1;
        if (kc + 1 < K / 32) { stage(kc + 1, buf ^ 1); wait_async3(); }
        else                 { wait_async0(); }
        __syncthreads();
        const _Float16* At = smem + buf * 4096;
        const _Float16* Bt = smem + 8192 + buf * 2048;
        v16h af = load_af(At + wave * 16 * 32, 32);
        #pragma unroll
        for (int nt = 0; nt < 4; nt++) {
            v16h bf = load_bf(Bt + nt * 16 * 32, 32);
            acc[nt] = wmma32(af, bf, acc[nt]);
        }
        __syncthreads();
    }

    #pragma unroll
    for (int nt = 0; nt < 4; nt++) {
        int c = col0 + nt * 16 + nn;
        #pragma unroll
        for (int r = 0; r < 8; r++) {
            int g = row0 + wave * 16 + r + 8 * half;
            out[(size_t)g * Esz + c] = acc[nt][r];
        }
    }
}

extern "C" void kernel_launch(void* const* d_in, const int* in_sizes, int n_in,
                              void* d_out, int out_size, void* d_ws, size_t ws_size,
                              hipStream_t stream) {
    const float* X     = (const float*)d_in[0];   // [4,2048,1024]
    const float* W_qkv = (const float*)d_in[1];   // [1024,3072]
    const float* b_qkv = (const float*)d_in[2];   // [3072]
    const float* W_out = (const float*)d_in[3];   // [1024,1024]
    const float* b_out = (const float*)d_in[4];   // [1024]
    float* out = (float*)d_out;

    char* ws = (char*)d_ws;
    size_t off = 0;
    auto alloc = [&](size_t bytes) { char* p = ws + off; off += (bytes + 255) & ~(size_t)255; return p; };
    _Float16* Xh     = (_Float16*)alloc((size_t)Bsz * Lsz * Esz * 2);          // 16 MB
    _Float16* Wt_qkv = (_Float16*)alloc((size_t)3072 * Esz * 2);               //  6 MB
    _Float16* Wt_out = (_Float16*)alloc((size_t)Esz * Esz * 2);                //  2 MB
    _Float16* Qh     = (_Float16*)alloc((size_t)Bsz * Hsz * Lsz * Dsz * 2);    // 16 MB
    _Float16* Kh     = (_Float16*)alloc((size_t)Bsz * Hsz * Lsz * Dsz * 2);    // 16 MB
    _Float16* Vt     = (_Float16*)alloc((size_t)Bsz * Hsz * Dsz * Lsz * 2);    // 16 MB
    _Float16* attnH  = (_Float16*)alloc((size_t)Bsz * Lsz * Hsz * Dsz * 2);    // 16 MB
    (void)ws_size; (void)in_sizes; (void)n_in; (void)out_size;

    int nX = Bsz * Lsz * Esz;
    cvt_f32_f16<<<(nX + 255) / 256, 256, 0, stream>>>(X, Xh, nX);
    transpose_to_f16<<<(Esz * 3072 + 255) / 256, 256, 0, stream>>>(W_qkv, Wt_qkv, Esz, 3072);
    transpose_to_f16<<<(Esz * Esz + 255) / 256, 256, 0, stream>>>(W_out, Wt_out, Esz, Esz);

    // 64 row-tiles x 48 col-tiles = 3072 blocks (128x64 tile per block)
    qkv_gemm<<<3072, 256, 0, stream>>>(Xh, Wt_qkv, b_qkv, Qh, Kh, Vt);

    // 4*16*128 = 8192 wave-tiles / 8 = 1024 blocks
    attn_kernel<<<1024, 256, 0, stream>>>(Qh, Kh, Vt, attnH);

    // 64 row-tiles x 16 col-tiles = 1024 blocks
    out_gemm<<<1024, 256, 0, stream>>>(attnH, Wt_out, b_out, out);
}
